// HiddenMarkovModel_40029095198930
// MI455X (gfx1250) — matrix operand order repaired and checked by reading, not compiled, not run on hardware
//
#include <hip/hip_runtime.h>
#include <hip/hip_bf16.h>

typedef __attribute__((ext_vector_type(16))) __bf16 bf16x16;
typedef __attribute__((ext_vector_type(8)))  __bf16 bf16x8;
typedef __attribute__((ext_vector_type(4)))  __bf16 bf16x4;
typedef __attribute__((ext_vector_type(8)))  float  v8f;

#define WMMA_BF16(A, B, C) \
  __builtin_amdgcn_wmma_f32_16x16x32_bf16(false, (A), false, (B), (short)0, (C), false, false)

// ds_swizzle group-of-32 xor pattern: new_lane = (lane & 0x1f) ^ mask
// masks 1/2/4/8 keep exchanges within each 16-lane half (== shfl_xor width 16).
#define DS_SWZ(x, imm) __int_as_float(__builtin_amdgcn_ds_swizzle(__float_as_int(x), (imm)))

__device__ __forceinline__ float redmax16(float x) {
  x = fmaxf(x, DS_SWZ(x, (1 << 10) | 0x1f));
  x = fmaxf(x, DS_SWZ(x, (2 << 10) | 0x1f));
  x = fmaxf(x, DS_SWZ(x, (4 << 10) | 0x1f));
  x = fmaxf(x, DS_SWZ(x, (8 << 10) | 0x1f));
  return x;
}
__device__ __forceinline__ float redsum16(float x) {
  x += DS_SWZ(x, (1 << 10) | 0x1f);
  x += DS_SWZ(x, (2 << 10) | 0x1f);
  x += DS_SWZ(x, (4 << 10) | 0x1f);
  x += DS_SWZ(x, (8 << 10) | 0x1f);
  return x;
}

constexpr int BATCH = 64;
constexpr int DIM   = 64;   // D
constexpr int NST   = 64;   // N states
constexpr float LOG_2PI = 1.8378770664093453f;

// ---------------------------------------------------------------------------
// Prep: softmax(transition) transposed (bf16), emission coefficients (bf16),
// bias + log_pi (f32), and zero the scalar output. One tiny workgroup.
// ---------------------------------------------------------------------------
__global__ void hmm_prep(const float* __restrict__ trans,
                         const float* __restrict__ priors,
                         const float* __restrict__ means,
                         const float* __restrict__ lvars,
                         __bf16* __restrict__ At,    // [n][m] = softmax(trans)[m][n]
                         __bf16* __restrict__ W1t,   // [n][d] = mu*inv_var
                         __bf16* __restrict__ W2t,   // [n][d] = -0.5*inv_var
                         float*  __restrict__ bias,  // [n]
                         float*  __restrict__ logpi, // [n]
                         float*  __restrict__ out)
{
  const int i = threadIdx.x;            // 0..63, one row / state each
  if (i == 0) out[0] = 0.0f;

  float m = -1e30f;
  for (int j = 0; j < NST; ++j) m = fmaxf(m, trans[i * NST + j]);
  float s = 0.0f;
  float e[NST];
  for (int j = 0; j < NST; ++j) { e[j] = __expf(trans[i * NST + j] - m); s += e[j]; }
  const float inv_s = 1.0f / s;
  for (int j = 0; j < NST; ++j) At[j * NST + i] = (__bf16)(e[j] * inv_s);

  float slv = 0.0f, sm2 = 0.0f;
  for (int d = 0; d < DIM; ++d) {
    const float lv = lvars[i * DIM + d];
    const float iv = __expf(-lv);
    const float mu = means[i * DIM + d];
    W1t[i * DIM + d] = (__bf16)(mu * iv);
    W2t[i * DIM + d] = (__bf16)(-0.5f * iv);
    slv += lv;
    sm2 += mu * mu * iv;
  }
  bias[i] = -0.5f * ((float)DIM * LOG_2PI + slv) - 0.5f * sm2;

  float pm = -1e30f;
  for (int j = 0; j < NST; ++j) pm = fmaxf(pm, priors[j]);
  float ps = 0.0f;
  for (int j = 0; j < NST; ++j) ps += __expf(priors[j] - pm);
  logpi[i] = priors[i] - pm - __logf(ps);
}

// ---------------------------------------------------------------------------
// Bandwidth-bound pre-pass: Xp[b][t][0..63] = bf16(x), Xp[b][t][64..127] = bf16(x*x).
// Natural d-order == WMMA A-fragment chunk order (8 contiguous bf16 per half-frag).
// ---------------------------------------------------------------------------
__global__ void hmm_xprep(const float* __restrict__ X, __bf16* __restrict__ Xp,
                          long long nquads)
{
  const long long idx = (long long)blockIdx.x * blockDim.x + threadIdx.x;
  if (idx >= nquads) return;
  const long long bt = idx >> 4;      // which (b,t)
  const int dq = (int)(idx & 15);     // which group of 4 d's
  const float4 q = *(const float4*)(X + bt * DIM + dq * 4);
  bf16x4 v1, v2;
  v1[0] = (__bf16)q.x; v1[1] = (__bf16)q.y; v1[2] = (__bf16)q.z; v1[3] = (__bf16)q.w;
  v2[0] = (__bf16)(q.x * q.x); v2[1] = (__bf16)(q.y * q.y);
  v2[2] = (__bf16)(q.z * q.z); v2[3] = (__bf16)(q.w * q.w);
  *(bf16x4*)(Xp + bt * 128 + dq * 4)      = v1;
  *(bf16x4*)(Xp + bt * 128 + 64 + dq * 4) = v2;
}

// ---------------------------------------------------------------------------
// Forward scan. grid = 4 WGs (16 batch rows each), 4 waves (16-state tile each).
// PRE=true: A-fragments loaded directly from Xp (no in-loop conversion VALU).
// ---------------------------------------------------------------------------
union AF { bf16x16 v; bf16x8 h[2]; };

struct XBuf {        // only the members of the active path survive DCE
  float xf[32];      // PRE=false: raw f32, fragment-chunk order
  AF a1[2], a2[2];   // PRE=true : ready-to-use A-fragments (x and x^2)
};

template <bool PRE>
__global__ void __launch_bounds__(128, 1) hmm_forward(
    const float*  __restrict__ X,
    const __bf16* __restrict__ Xp,
    const __bf16* __restrict__ At,
    const __bf16* __restrict__ W1t,
    const __bf16* __restrict__ W2t,
    const float*  __restrict__ bias,
    const float*  __restrict__ logpi,
    float* __restrict__ out, int T)
{
  __shared__ alignas(32) __bf16 Ebuf[2][16 * NST];   // double-buffered exp(alpha - max)
  __shared__ alignas(16) float maxbuf[2][16][4];     // [parity][row][wave] -> b128 read
  __shared__ alignas(16) float sumb[16][4];
  __shared__ float rmaxb[16];

  const int tid  = threadIdx.x;
  const int wave = tid >> 5;
  const int lane = tid & 31;
  const int h0   = lane >> 4;       // lane half (K-offset selector)
  const int r    = lane & 15;       // A matrix row | B/C matrix column
  const int b0   = blockIdx.x * 16; // batch tile base
  const int n0   = wave * 16;       // state tile base

  // resident B-fragments: two 8-bf16 chunks per lane (k = 8*h0.., 16+8*h0..)
  bf16x16 w1f[2], w2f[2], atf[2];
#pragma unroll
  for (int f = 0; f < 2; ++f) {
    const int koff = 32 * f + 8 * h0;
    w1f[f] = __builtin_shufflevector(*(const bf16x8*)(W1t + (n0 + r) * DIM + koff), *(const bf16x8*)(W1t + (n0 + r) * DIM + koff + 16), 0, 1, 2, 3, 4, 5, 6, 7, 8, 9, 10, 11, 12, 13, 14, 15);
    w2f[f] = __builtin_shufflevector(*(const bf16x8*)(W2t + (n0 + r) * DIM + koff), *(const bf16x8*)(W2t + (n0 + r) * DIM + koff + 16), 0, 1, 2, 3, 4, 5, 6, 7, 8, 9, 10, 11, 12, 13, 14, 15);
    atf[f] = __builtin_shufflevector(*(const bf16x8*)(At  + (n0 + r) * NST + koff), *(const bf16x8*)(At  + (n0 + r) * NST + koff + 16), 0, 1, 2, 3, 4, 5, 6, 7, 8, 9, 10, 11, 12, 13, 14, 15);
  }
  const float bias_n = bias[n0 + r];
  const float lpi_n  = logpi[n0 + r];

  const float*  xrow  = X  + (size_t)(b0 + r) * (size_t)T * DIM;  // this lane's batch row
  const __bf16* xprow = Xp + (size_t)(b0 + r) * (size_t)T * 128;

  auto load_t = [&](XBuf& bUf, int t) {
    if constexpr (PRE) {
      const __bf16* p = xprow + (size_t)t * 128 + 8 * h0;
      bUf.a1[0].h[0] = *(const bf16x8*)(p);
      bUf.a1[0].h[1] = *(const bf16x8*)(p + 16);
      bUf.a1[1].h[0] = *(const bf16x8*)(p + 32);
      bUf.a1[1].h[1] = *(const bf16x8*)(p + 48);
      bUf.a2[0].h[0] = *(const bf16x8*)(p + 64);
      bUf.a2[0].h[1] = *(const bf16x8*)(p + 80);
      bUf.a2[1].h[0] = *(const bf16x8*)(p + 96);
      bUf.a2[1].h[1] = *(const bf16x8*)(p + 112);
    } else {
      const float* pt = xrow + (size_t)t * DIM;
#pragma unroll
      for (int c = 0; c < 4; ++c) {
        const float4 q0 = *(const float4*)(pt + 16 * c + 8 * h0);
        const float4 q1 = *(const float4*)(pt + 16 * c + 8 * h0 + 4);
        bUf.xf[c*8+0]=q0.x; bUf.xf[c*8+1]=q0.y; bUf.xf[c*8+2]=q0.z; bUf.xf[c*8+3]=q0.w;
        bUf.xf[c*8+4]=q1.x; bUf.xf[c*8+5]=q1.y; bUf.xf[c*8+6]=q1.z; bUf.xf[c*8+7]=q1.w;
      }
    }
  };

  auto emis = [&](const XBuf& bUf) -> v8f {
    v8f em = {};
    if constexpr (PRE) {
      em = WMMA_BF16(bUf.a2[1].v, w2f[1], em);
      em = WMMA_BF16(bUf.a2[0].v, w2f[0], em);
      em = WMMA_BF16(bUf.a1[1].v, w1f[1], em);
      em = WMMA_BF16(bUf.a1[0].v, w1f[0], em);
    } else {
      bf16x16 a1[2], a2[2];
#pragma unroll
      for (int f = 0; f < 2; ++f)
#pragma unroll
        for (int i = 0; i < 16; ++i) {
          const int v = i >> 1;
          const float x = bUf.xf[(2 * f + (v >> 2)) * 8 + ((v & 3) << 1) + (i & 1)];
          a1[f][i] = (__bf16)x;
          a2[f][i] = (__bf16)(x * x);
        }
      em = WMMA_BF16(a2[1], w2f[1], em);
      em = WMMA_BF16(a2[0], w2f[0], em);
      em = WMMA_BF16(a1[1], w1f[1], em);
      em = WMMA_BF16(a1[0], w1f[0], em);
    }
    return em;
  };

  float alpha[8];   // C-layout: lane holds rows 8*h0+v, column n0+r
  XBuf xa, xb;

  // ---- t = 0: alpha0 = log_pi + logB(t=0) ----
  load_t(xb, 0);
  {
    const v8f em = emis(xb);
#pragma unroll
    for (int v = 0; v < 8; ++v) alpha[v] = lpi_n + em[v] + bias_n;
  }
  load_t(xa, 1);   // prefetch x(t=1)

  auto do_step = [&](int t, XBuf& xc, XBuf& xn) {
    const int p = t & 1;
    const v8f em = emis(xc);                    // independent of alpha: overlaps chain
    const int tn = (t + 1 < T) ? (t + 1) : t;
    load_t(xn, tn);                             // prefetch x(t+1)

    // --- row max of alpha over all 64 states (ds_swizzle + one b128 LDS combine) ---
    float pm[8];
#pragma unroll
    for (int v = 0; v < 8; ++v) pm[v] = redmax16(alpha[v]);
    if (r == 0) {
#pragma unroll
      for (int v = 0; v < 8; ++v) maxbuf[p][8 * h0 + v][wave] = pm[v];
    }
    __syncthreads();
    float rmax[8];
#pragma unroll
    for (int v = 0; v < 8; ++v) {
      const float4 q = *(const float4*)&maxbuf[p][8 * h0 + v][0];
      rmax[v] = fmaxf(fmaxf(q.x, q.y), fmaxf(q.z, q.w));
    }
    // --- E = exp(alpha - rowmax) -> LDS (bf16, row-major [16 x 64]) ---
#pragma unroll
    for (int v = 0; v < 8; ++v) {
      const float e = __expf(alpha[v] - rmax[v]);
      Ebuf[p][(8 * h0 + v) * NST + n0 + r] = (__bf16)e;
    }
    __syncthreads();
    // --- reload E as A-fragments, transition matmul in probability domain ---
    AF ea0, ea1;
    ea0.h[0] = *(const bf16x8*)&Ebuf[p][r * NST + 8 * h0];
    ea0.h[1] = *(const bf16x8*)&Ebuf[p][r * NST + 16 + 8 * h0];
    ea1.h[0] = *(const bf16x8*)&Ebuf[p][r * NST + 32 + 8 * h0];
    ea1.h[1] = *(const bf16x8*)&Ebuf[p][r * NST + 48 + 8 * h0];
    v8f s = {};
    s = WMMA_BF16(ea1.v, atf[1], s);
    s = WMMA_BF16(ea0.v, atf[0], s);
#pragma unroll
    for (int v = 0; v < 8; ++v)
      alpha[v] = em[v] + bias_n + rmax[v] + __logf(s[v]);
  };

  int t = 1;
  for (; t + 1 < T; t += 2) { do_step(t, xa, xb); do_step(t + 1, xb, xa); }
  if (t < T) do_step(t, xa, xb);

  // ---- final: sum_b logsumexp_n(alpha_T[b,n]) ----
  {
    float pm[8];
#pragma unroll
    for (int v = 0; v < 8; ++v) pm[v] = redmax16(alpha[v]);
    if (r == 0) {
#pragma unroll
      for (int v = 0; v < 8; ++v) maxbuf[0][8 * h0 + v][wave] = pm[v];
    }
    __syncthreads();
    float rmax[8];
#pragma unroll
    for (int v = 0; v < 8; ++v) {
      const float4 q = *(const float4*)&maxbuf[0][8 * h0 + v][0];
      rmax[v] = fmaxf(fmaxf(q.x, q.y), fmaxf(q.z, q.w));
    }
    float sv[8];
#pragma unroll
    for (int v = 0; v < 8; ++v) sv[v] = redsum16(__expf(alpha[v] - rmax[v]));
    if (r == 0) {
#pragma unroll
      for (int v = 0; v < 8; ++v) {
        sumb[8 * h0 + v][wave] = sv[v];
        if (wave == 0) rmaxb[8 * h0 + v] = rmax[v];
      }
    }
    __syncthreads();
    if (tid < 16) {
      const float4 q = *(const float4*)&sumb[tid][0];
      float lp = __logf((q.x + q.y) + (q.z + q.w)) + rmaxb[tid];
      lp = redsum16(lp);
      if (tid == 0) atomicAdd(out, lp);
    }
  }
}

// ---------------------------------------------------------------------------
extern "C" void kernel_launch(void* const* d_in, const int* in_sizes, int n_in,
                              void* d_out, int out_size, void* d_ws, size_t ws_size,
                              hipStream_t stream)
{
  const float* X      = (const float*)d_in[0];   // [B,T,D] f32
  const float* trans  = (const float*)d_in[1];   // [N,N]   f32
  const float* priors = (const float*)d_in[2];   // [N]     f32
  const float* means  = (const float*)d_in[3];   // [N,D]   f32
  const float* lvars  = (const float*)d_in[4];   // [N,D]   f32
  float* out = (float*)d_out;                    // scalar f32
  const int T = in_sizes[0] / (BATCH * DIM);

  char* ws = (char*)d_ws;
  __bf16* At    = (__bf16*)(ws);                 // 8 KB
  __bf16* W1t   = (__bf16*)(ws + 8192);          // 8 KB
  __bf16* W2t   = (__bf16*)(ws + 16384);         // 8 KB
  float*  bias  = (float*)(ws + 24576);          // 256 B
  float*  logpi = (float*)(ws + 24576 + 256);    // 256 B
  __bf16* Xp    = (__bf16*)(ws + 32768);         // optional: B*T*128 bf16 (~134 MB)

  hmm_prep<<<dim3(1), dim3(64), 0, stream>>>(trans, priors, means, lvars,
                                             At, W1t, W2t, bias, logpi, out);

  const size_t xp_bytes = (size_t)BATCH * (size_t)T * 128 * sizeof(__bf16);
  if (ws_size >= 32768 + xp_bytes) {
    const long long nquads = (long long)BATCH * T * (DIM / 4);
    const int blocks = (int)((nquads + 255) / 256);
    hmm_xprep<<<dim3(blocks), dim3(256), 0, stream>>>(X, Xp, nquads);
    hmm_forward<true><<<dim3(4), dim3(128), 0, stream>>>(X, Xp, At, W1t, W2t,
                                                         bias, logpi, out, T);
  } else {
    hmm_forward<false><<<dim3(4), dim3(128), 0, stream>>>(X, Xp, At, W1t, W2t,
                                                          bias, logpi, out, T);
  }
}
